// ReDrafterHead_61160334295638
// MI455X (gfx1250) — matrix operand order repaired and checked by reading, not compiled, not run on hardware
//
#include <hip/hip_runtime.h>

// ---------------------------------------------------------------------------
// ReDrafter draft head for MI455X (gfx1250, wave32, WMMA).
// GEMMs: v_wmma_f32_16x16x32_bf16, f32 operands converted to bf16 in-register
// (v_cvt_pk_bf16_f32), f32 accumulation. Each wave computes a 16x64 output
// strip (1 M-tile x 4 N-tiles). Per K-step: 20 global_load_b128 issued via
// inline asm into distinct register quads, ONE operand-tied s_wait_loadcnt 0,
// all fragments converted, then an empty asm tie forces the 4 B fragments
// into DISJOINT register octets so the 4 WMMAs issue with no WAR v_nops.
// Serial K=4 draft chain handled by stream-ordered launches (graph-safe).
// ---------------------------------------------------------------------------

typedef __attribute__((ext_vector_type(16))) __bf16 v16bf;
typedef __attribute__((ext_vector_type(8)))  float  v8f;
typedef __attribute__((ext_vector_type(4)))  float  f4;

#define B_SZ   64
#define D_SZ   512
#define H_SZ   4096
#define V_SZ   32000
#define KSTEPS 4
#define TD_SZ  1536   // 3*D

// 4 x b128 from p at byte offsets {0,16,32,48}  (B fragment: 16 contiguous f32)
__device__ __forceinline__ void gload_b(const float* p, f4& r0, f4& r1, f4& r2, f4& r3) {
  asm volatile(
      "global_load_b128 %0, %4, off\n\t"
      "global_load_b128 %1, %4, off offset:16\n\t"
      "global_load_b128 %2, %4, off offset:32\n\t"
      "global_load_b128 %3, %4, off offset:48"
      : "=&v"(r0), "=&v"(r1), "=&v"(r2), "=&v"(r3)
      : "v"(p));
}

// 4 x b128 from p at byte offsets {0,16,64,80}  (A fragment: two 8-f32 chunks 64B apart)
__device__ __forceinline__ void gload_a(const float* p, f4& r0, f4& r1, f4& r2, f4& r3) {
  asm volatile(
      "global_load_b128 %0, %4, off\n\t"
      "global_load_b128 %1, %4, off offset:16\n\t"
      "global_load_b128 %2, %4, off offset:64\n\t"
      "global_load_b128 %3, %4, off offset:80"
      : "=&v"(r0), "=&v"(r1), "=&v"(r2), "=&v"(r3)
      : "v"(p));
}

__device__ __forceinline__ v16bf pack_bf16(f4 p0, f4 p1, f4 p2, f4 p3) {
  v16bf r;
  #pragma unroll
  for (int i = 0; i < 4; ++i) {
    r[i]      = (__bf16)p0[i];
    r[4 + i]  = (__bf16)p1[i];
    r[8 + i]  = (__bf16)p2[i];
    r[12 + i] = (__bf16)p3[i];
  }
  return r;
}

// Out[m,n] = sum_k A[m,k]*W[n,k] + bias[n]
// A-frag (16x32 bf16): lane<16 -> elems 0..15 = K+0..15;
//                      lane>=16 -> elems 0..15 = K+16..31 (same order as B).
// B-frag (32x16 bf16): N = lane&15, half selects K+0..15 vs K+16..31.
// C/D (16x16 f32): N = lane&15, M = (lane>>4)*8 + r.
__global__ void gemm_bf16_wmma(const float* __restrict__ A, int lda,
                               const float* __restrict__ W, int ldw,
                               const float* __restrict__ bias,
                               float* __restrict__ Out, long ldo,
                               int mTiles, int nGroups, int ksteps)
{
  const int wavesPerBlock = blockDim.x >> 5;
  const int wave = blockIdx.x * wavesPerBlock + ((int)threadIdx.x >> 5);
  if (wave >= mTiles * nGroups) return;            // wave-uniform: EXEC stays all-1
  const int tileM = wave % mTiles;
  const int group = wave / mTiles;                 // 4 consecutive N tiles

  const int lane = threadIdx.x & 31;
  const int half = lane >> 4;                      // 0 or 1
  const int l15  = lane & 15;

  const float* __restrict__ arow = A + (long)(tileM * 16 + l15) * lda + half * 16;
  const float* w0 = W + (long)((group * 4 + 0) * 16 + l15) * ldw + half * 16;
  const float* w1 = W + (long)((group * 4 + 1) * 16 + l15) * ldw + half * 16;
  const float* w2 = W + (long)((group * 4 + 2) * 16 + l15) * ldw + half * 16;
  const float* w3 = W + (long)((group * 4 + 3) * 16 + l15) * ldw + half * 16;

  v8f acc0, acc1, acc2, acc3;
  #pragma unroll
  for (int i = 0; i < 8; ++i) { acc0[i] = 0.f; acc1[i] = 0.f; acc2[i] = 0.f; acc3[i] = 0.f; }

  for (int kk = 0; kk < ksteps; ++kk) {
    const int k0 = kk << 5;

    // ---- issue all 20 b128 loads back-to-back (volatile order) ----
    f4 a0, a1, a2, a3;
    f4 b00, b01, b02, b03, b10, b11, b12, b13;
    f4 b20, b21, b22, b23, b30, b31, b32, b33;
    gload_b(arow + k0, a0, a1, a2, a3);
    gload_b(w0 + k0, b00, b01, b02, b03);
    gload_b(w1 + k0, b10, b11, b12, b13);
    gload_b(w2 + k0, b20, b21, b22, b23);
    gload_b(w3 + k0, b30, b31, b32, b33);

    // ---- ONE wait; ties force every consumer after this point ----
    asm volatile("s_wait_loadcnt 0x0"
        : "+v"(a0), "+v"(a1), "+v"(a2), "+v"(a3),
          "+v"(b00), "+v"(b01), "+v"(b02), "+v"(b03),
          "+v"(b10), "+v"(b11), "+v"(b12), "+v"(b13),
          "+v"(b20), "+v"(b21), "+v"(b22), "+v"(b23),
          "+v"(b30), "+v"(b31), "+v"(b32), "+v"(b33));

    // ---- convert ALL fragments ----
    v16bf a   = pack_bf16(a0, a1, a2, a3);
    v16bf bf0 = pack_bf16(b00, b01, b02, b03);
    v16bf bf1 = pack_bf16(b10, b11, b12, b13);
    v16bf bf2 = pack_bf16(b20, b21, b22, b23);
    v16bf bf3 = pack_bf16(b30, b31, b32, b33);

    // Empty tie: forces a, bf0..bf3 simultaneously live -> disjoint register
    // octets, so the 4 WMMAs below issue back-to-back without WAR v_nops.
    asm volatile(""
        : "+v"(a), "+v"(bf0), "+v"(bf1), "+v"(bf2), "+v"(bf3));

    // ---- 4 independent WMMAs back-to-back ----
    acc0 = __builtin_amdgcn_wmma_f32_16x16x32_bf16(false, a, false, bf0, (short)0, acc0, false, false);
    acc1 = __builtin_amdgcn_wmma_f32_16x16x32_bf16(false, a, false, bf1, (short)0, acc1, false, false);
    acc2 = __builtin_amdgcn_wmma_f32_16x16x32_bf16(false, a, false, bf2, (short)0, acc2, false, false);
    acc3 = __builtin_amdgcn_wmma_f32_16x16x32_bf16(false, a, false, bf3, (short)0, acc3, false, false);
  }

  const int mBase = tileM * 16 + half * 8;
  v8f accs[4] = {acc0, acc1, acc2, acc3};
  #pragma unroll
  for (int t = 0; t < 4; ++t) {
    const int nOut = (group * 4 + t) * 16 + l15;
    const float bv = bias ? bias[nOut] : 0.0f;
    float* __restrict__ col = Out + nOut;
    #pragma unroll
    for (int r = 0; r < 8; ++r)
      col[(long)(mBase + r) * ldo] = accs[t][r] + bv;
  }
}

// x := 0, argmax keys := 0
__global__ void init_state(float* __restrict__ x, unsigned long long* __restrict__ keys) {
  const int i = blockIdx.x * blockDim.x + threadIdx.x;
  if (i < B_SZ * D_SZ) x[i] = 0.0f;
  if (i < B_SZ) keys[i] = 0ull;
}

__global__ void reset_keys(unsigned long long* __restrict__ keys) {
  if (threadIdx.x < B_SZ) keys[threadIdx.x] = 0ull;
}

__global__ void copy_vec(const float* __restrict__ src, float* __restrict__ dst, int n) {
  const int i = blockIdx.x * blockDim.x + threadIdx.x;
  if (i < n) dst[i] = src[i];
}

// GRU gate fusion (PyTorch order r,z,n): h := (1-z)*n + z*h
__global__ void gru_combine(const float* __restrict__ gi,
                            const float* __restrict__ gh,
                            float* __restrict__ h)
{
  const int idx = blockIdx.x * blockDim.x + threadIdx.x;
  if (idx >= B_SZ * D_SZ) return;
  const int b = idx >> 9;            // /512
  const int d = idx & (D_SZ - 1);
  const float* gib = gi + b * TD_SZ;
  const float* ghb = gh + b * TD_SZ;
  const float ir = gib[d],             hr = ghb[d];
  const float iz = gib[D_SZ + d],      hz = ghb[D_SZ + d];
  const float in_ = gib[2 * D_SZ + d], hn = ghb[2 * D_SZ + d];
  const float r = 1.0f / (1.0f + __expf(-(ir + hr)));
  const float z = 1.0f / (1.0f + __expf(-(iz + hz)));
  const float n = tanhf(in_ + r * hn);
  h[idx] = (1.0f - z) * n + z * h[idx];
}

// Deterministic argmax over V per batch row via monotonic packed key + atomicMax.
// key = (orderable(float) << 32) | (V - v)  -> ties resolve to smallest v.
__global__ void vocab_argmax(const float* __restrict__ logits, long ldl,
                             unsigned long long* __restrict__ keys)
{
  const int b = blockIdx.y;
  const float* __restrict__ row = logits + (long)b * ldl;
  const int base = blockIdx.x * 1000;          // 32 chunks * 1000 = 32000

  unsigned long long best = 0ull;
  for (int i = threadIdx.x; i < 1000; i += blockDim.x) {
    const int v = base + i;
    unsigned u = __float_as_uint(row[v]);
    u = (u & 0x80000000u) ? ~u : (u | 0x80000000u);
    const unsigned long long key =
        ((unsigned long long)u << 32) | (unsigned)(V_SZ - v);
    best = (key > best) ? key : best;
  }

  __shared__ unsigned long long sm[256];
  sm[threadIdx.x] = best;
  __syncthreads();
  for (int s = 128; s > 0; s >>= 1) {
    if ((int)threadIdx.x < s) {
      const unsigned long long o = sm[threadIdx.x + s];
      if (o > sm[threadIdx.x]) sm[threadIdx.x] = o;
    }
    __syncthreads();
  }
  if (threadIdx.x == 0) atomicMax(&keys[b], sm[0]);
}

// x[b,:] = embed[argmax_b, :]
__global__ void gather_embed(const unsigned long long* __restrict__ keys,
                             const float* __restrict__ embed,
                             float* __restrict__ x)
{
  const int b = blockIdx.x;
  const int idx = V_SZ - (int)(unsigned)(keys[b] & 0xFFFFFFFFull);
  const float* __restrict__ src = embed + (long)idx * D_SZ;
  for (int d = threadIdx.x; d < D_SZ; d += blockDim.x)
    x[b * D_SZ + d] = src[d];
}

extern "C" void kernel_launch(void* const* d_in, const int* in_sizes, int n_in,
                              void* d_out, int out_size, void* d_ws, size_t ws_size,
                              hipStream_t stream) {
  (void)in_sizes; (void)n_in; (void)out_size; (void)ws_size;

  const float* hidden    = (const float*)d_in[0];   // [64,4096]
  const float* in_proj_w = (const float*)d_in[1];   // [512,4096]
  const float* in_proj_b = (const float*)d_in[2];   // [512]
  const float* w_ih0     = (const float*)d_in[3];   // [1536,512]
  const float* w_hh0     = (const float*)d_in[4];
  const float* b_ih0     = (const float*)d_in[5];
  const float* b_hh0     = (const float*)d_in[6];
  const float* w_ih1     = (const float*)d_in[7];
  const float* w_hh1     = (const float*)d_in[8];
  const float* b_ih1     = (const float*)d_in[9];
  const float* b_hh1     = (const float*)d_in[10];
  const float* embed     = (const float*)d_in[11];  // [32000,512]
  const float* out_w     = (const float*)d_in[12];  // [32000,512]
  float* out = (float*)d_out;                       // [64,4,32000]

  float* ws = (float*)d_ws;
  float* h0 = ws;                  // [64,512]
  float* h1 = ws + 32768;          // [64,512]
  float* x  = ws + 65536;          // [64,512]
  float* gi = ws + 98304;          // [64,1536]
  float* gh = ws + 196608;         // [64,1536]
  unsigned long long* keys = (unsigned long long*)(ws + 294912);  // [64]

  // x=0, keys=0
  init_state<<<128, 256, 0, stream>>>(x, keys);

  // h0 = hidden @ in_proj_w^T + b   (M=64, N=512, K=4096); then h1 = h0
  gemm_bf16_wmma<<<4, 256, 0, stream>>>(hidden, H_SZ, in_proj_w, H_SZ, in_proj_b,
                                        h0, D_SZ, 4, (D_SZ / 16) / 4, H_SZ / 32);
  copy_vec<<<128, 256, 0, stream>>>(h0, h1, B_SZ * D_SZ);

  for (int k = 0; k < KSTEPS; ++k) {
    // Layer 0: gi = x @ w_ih0^T + b_ih0 ; gh = h0 @ w_hh0^T + b_hh0
    gemm_bf16_wmma<<<12, 256, 0, stream>>>(x,  D_SZ, w_ih0, D_SZ, b_ih0,
                                           gi, TD_SZ, 4, (TD_SZ / 16) / 4, D_SZ / 32);
    gemm_bf16_wmma<<<12, 256, 0, stream>>>(h0, D_SZ, w_hh0, D_SZ, b_hh0,
                                           gh, TD_SZ, 4, (TD_SZ / 16) / 4, D_SZ / 32);
    gru_combine<<<128, 256, 0, stream>>>(gi, gh, h0);

    // Layer 1: gi = h0 @ w_ih1^T + b_ih1 ; gh = h1 @ w_hh1^T + b_hh1
    gemm_bf16_wmma<<<12, 256, 0, stream>>>(h0, D_SZ, w_ih1, D_SZ, b_ih1,
                                           gi, TD_SZ, 4, (TD_SZ / 16) / 4, D_SZ / 32);
    gemm_bf16_wmma<<<12, 256, 0, stream>>>(h1, D_SZ, w_hh1, D_SZ, b_hh1,
                                           gh, TD_SZ, 4, (TD_SZ / 16) / 4, D_SZ / 32);
    gru_combine<<<128, 256, 0, stream>>>(gi, gh, h1);

    // logits[b, k*V + v] : Out base = out + k*V, row stride K*V
    gemm_bf16_wmma<<<250, 256, 0, stream>>>(h1, D_SZ, out_w, D_SZ, nullptr,
                                            out + (long)k * V_SZ, (long)KSTEPS * V_SZ,
                                            4, (V_SZ / 16) / 4, D_SZ / 32);

    // argmax -> keys, then x = embed[argmax]
    reset_keys<<<1, 64, 0, stream>>>(keys);
    vocab_argmax<<<dim3(32, 64), 256, 0, stream>>>(out + (long)k * V_SZ,
                                                   (long)KSTEPS * V_SZ, keys);
    gather_embed<<<64, 256, 0, stream>>>(keys, embed, x);
  }
}
